// RNNBlock_65738769433283
// MI455X (gfx1250) — compile-verified, matrix operand order checked by reading, not run on hardware
//
#include <hip/hip_runtime.h>
#include <hip/hip_bf16.h>

// Problem constants (reference: B=64, T=512, D=1024, H=1024)
#define RNN_B 64
#define RNN_T 512
#define RNN_D 1024
#define RNN_H 1024

// LDS tile row pitch in 16-bit elements: 32 data + 8 pad = 80 bytes/row,
// 16B aligned for ds_load_b128 / async b128 copies, padded for banks.
#define LDS_PITCH 40

typedef __bf16 bf16_t;
typedef bf16_t v8bf  __attribute__((ext_vector_type(8)));
typedef bf16_t v16bf __attribute__((ext_vector_type(16)));
typedef float  v8f   __attribute__((ext_vector_type(8)));

// Exact parameter type of the async-to-LDS builtin (from hipcc diagnostic):
// pointer to GCC-style vector of 4 ints, addrspace(1) src / addrspace(3) dst.
typedef int async_v4i __attribute__((vector_size(16)));
#define AS1 __attribute__((address_space(1)))
#define AS3 __attribute__((address_space(3)))

// gfx1250 async global->LDS copy path (GLOBAL_LOAD_ASYNC_TO_LDS_B128),
// guarded so a toolchain without the builtin still compiles (reg-staged path).
#if defined(__gfx1250__) && defined(__has_builtin)
#if __has_builtin(__builtin_amdgcn_global_load_async_to_lds_b128)
#define USE_ASYNC_LDS 1
#endif
#endif
#ifndef USE_ASYNC_LDS
#define USE_ASYNC_LDS 0
#endif

__device__ __forceinline__ void wait_async0() {
#if defined(__gfx1250__)
#if defined(__has_builtin)
#if __has_builtin(__builtin_amdgcn_s_wait_asynccnt)
  __builtin_amdgcn_s_wait_asynccnt(0);
#else
  asm volatile("s_wait_asynccnt 0x0" ::: "memory");
#endif
#else
  asm volatile("s_wait_asynccnt 0x0" ::: "memory");
#endif
#endif
}

// ---------------------------------------------------------------------------
// f32 -> bf16 round-to-nearest-even, pure bit manipulation
// ---------------------------------------------------------------------------
__device__ __forceinline__ unsigned short f2bf(float f) {
  unsigned int u = __float_as_uint(f);
  u += 0x7fffu + ((u >> 16) & 1u);
  return (unsigned short)(u >> 16);
}

// ---------------------------------------------------------------------------
// Convert inputs to bf16 in workspace; zero initial hidden state h0.
// ---------------------------------------------------------------------------
__global__ void convert_kernel(const float* __restrict__ X,
                               const float* __restrict__ We,
                               const float* __restrict__ Wh,
                               unsigned short* __restrict__ Xb,
                               unsigned short* __restrict__ Web,
                               unsigned short* __restrict__ Whb,
                               unsigned short* __restrict__ h0) {
  const long long stride = (long long)gridDim.x * blockDim.x;
  const long long tid = (long long)blockIdx.x * blockDim.x + threadIdx.x;
  const long long nX = (long long)RNN_B * RNN_T * RNN_D;
  const long long nW = (long long)RNN_H * RNN_D;
  const long long nH = (long long)RNN_B * RNN_H;
  for (long long i = tid; i < nX; i += stride) Xb[i] = f2bf(X[i]);
  for (long long i = tid; i < nW; i += stride) Web[i] = f2bf(We[i]);
  for (long long i = tid; i < nW; i += stride) Whb[i] = f2bf(Wh[i]);
  for (long long i = tid; i < nH; i += stride) h0[i] = 0;
}

// ---------------------------------------------------------------------------
// Staging helpers: move a (ITERS*THREADS/4) x 32 bf16 tile (row stride ld)
// into LDS rows of LDS_PITCH. Chunk = 16 bytes. Fully unrolled.
// ---------------------------------------------------------------------------
#if USE_ASYNC_LDS
template<int THREADS, int ITERS>
__device__ __forceinline__ void stage_async(const unsigned short* __restrict__ src,
                                            int ld,
                                            unsigned short* __restrict__ dst) {
#pragma unroll
  for (int i = 0; i < ITERS; ++i) {
    const int c = threadIdx.x + i * THREADS;
    const int row = c >> 2;
    const int sub = c & 3;
    __builtin_amdgcn_global_load_async_to_lds_b128(
        (AS1 async_v4i*)(src + (long long)row * ld + sub * 8),
        (AS3 async_v4i*)(dst + row * LDS_PITCH + sub * 8),
        0, 0);
  }
}
#endif

template<int THREADS, int ITERS>
__device__ __forceinline__ void stage_gload(const unsigned short* __restrict__ src,
                                            int ld, v8bf (&regs)[ITERS]) {
#pragma unroll
  for (int i = 0; i < ITERS; ++i) {
    const int c = threadIdx.x + i * THREADS;
    const int row = c >> 2;
    const int sub = c & 3;
    const unsigned short* g = src + (long long)row * ld + sub * 8;
    regs[i] = *reinterpret_cast<const v8bf*>(g);
    __builtin_prefetch(g + 32, 0, 0);  // next K-chunk of this row
  }
}

template<int THREADS, int ITERS>
__device__ __forceinline__ void stage_sstore(unsigned short* __restrict__ dst,
                                             v8bf (&regs)[ITERS]) {
#pragma unroll
  for (int i = 0; i < ITERS; ++i) {
    const int c = threadIdx.x + i * THREADS;
    const int row = c >> 2;
    const int sub = c & 3;
    *reinterpret_cast<v8bf*>(dst + row * LDS_PITCH + sub * 8) = regs[i];
  }
}

// ---------------------------------------------------------------------------
// Load one 16x32 bf16 WMMA operand fragment from LDS.
// ISA 16-bit A/B layout: lane L holds row (L&15); lane-half (L>>4) selects
// K in [h*8, h*8+8) followed by [16+h*8, 16+h*8+8).
// ---------------------------------------------------------------------------
__device__ __forceinline__ v16bf frag_from_lds(const unsigned short* sm,
                                               int rowBase, int lane) {
  const int half = (lane >> 4) & 1;
  const int r = lane & 15;
  const unsigned short* p = sm + (rowBase + r) * LDS_PITCH + half * 8;
  v8bf lo = *reinterpret_cast<const v8bf*>(p);
  v8bf hi = *reinterpret_cast<const v8bf*>(p + 16);
  return __builtin_shufflevector(lo, hi, 0, 1, 2, 3, 4, 5, 6, 7,
                                 8, 9, 10, 11, 12, 13, 14, 15);
}

__device__ __forceinline__ v8f wmma_bf16(v16bf a, v16bf b, v8f c) {
#if defined(__gfx1250__)
  return __builtin_amdgcn_wmma_f32_16x16x32_bf16(
      false, a, false, b, (short)0, c, false, false);
#else
  return c;  // host pass: never executed on device
#endif
}

// Wave tile: 32(M) x 64(N) -> 2 A frags x 4 B frags, 8 WMMAs per 32-K chunk.
__device__ __forceinline__ void compute_tile(const unsigned short* __restrict__ smA,
                                             const unsigned short* __restrict__ smB,
                                             v8f (&acc)[2][4],
                                             int aRowBase, int bRowBase, int lane) {
  v16bf af[2], bfrag[4];
#pragma unroll
  for (int mt = 0; mt < 2; ++mt)
    af[mt] = frag_from_lds(smA, aRowBase + mt * 16, lane);
#pragma unroll
  for (int nt = 0; nt < 4; ++nt)
    bfrag[nt] = frag_from_lds(smB, bRowBase + nt * 16, lane);
#pragma unroll
  for (int mt = 0; mt < 2; ++mt)
#pragma unroll
    for (int nt = 0; nt < 4; ++nt)
      acc[mt][nt] = wmma_bf16(af[mt], bfrag[nt], acc[mt][nt]);
}

// ---------------------------------------------------------------------------
// Phase 1: xe[m, n] = sum_k Xb[m, k] * Web[n, k] + bias[n]   (m = b*T + t)
// Block: 256 threads = 8 waves (4x2), tile 128x128, K in 32-chunks,
// double-buffered LDS, one barrier per chunk, async/global staging overlapped
// with WMMA on the previous chunk. Output f32 in place into d_out.
// ---------------------------------------------------------------------------
__global__ __launch_bounds__(256)
void gemm_xe_kernel(const unsigned short* __restrict__ Xb,   // [B*T, D] bf16
                    const unsigned short* __restrict__ Web,  // [H, D]   bf16
                    const float* __restrict__ bias,          // [H]
                    float* __restrict__ out) {               // [B*T, H] f32
  __shared__ unsigned short sA[2][128 * LDS_PITCH];
  __shared__ unsigned short sB[2][128 * LDS_PITCH];

  const int mBase = blockIdx.x * 128;
  const int nBase = blockIdx.y * 128;
  const int wave  = threadIdx.x >> 5;
  const int lane  = threadIdx.x & 31;
  const int waveM = wave >> 1;  // 0..3
  const int waveN = wave & 1;   // 0..1
  const int aRowBase = waveM * 32;
  const int bRowBase = waveN * 64;

  const unsigned short* Ag = Xb  + (long long)mBase * RNN_D;
  const unsigned short* Bg = Web + (long long)nBase * RNN_D;

  v8f acc[2][4] = {};
  constexpr int NK = RNN_D / 32;

  // Prologue: stage K-chunk 0 into buffer 0
#if USE_ASYNC_LDS
  stage_async<256, 2>(Ag, RNN_D, sA[0]);
  stage_async<256, 2>(Bg, RNN_D, sB[0]);
  wait_async0();
#else
  {
    v8bf ra[2], rb[2];
    stage_gload<256, 2>(Ag, RNN_D, ra);
    stage_gload<256, 2>(Bg, RNN_D, rb);
    stage_sstore<256, 2>(sA[0], ra);
    stage_sstore<256, 2>(sB[0], rb);
  }
#endif
  __syncthreads();

  int cur = 0;
  for (int it = 1; it < NK; ++it) {
    const int kb = it * 32;
    const int nxt = cur ^ 1;
#if USE_ASYNC_LDS
    stage_async<256, 2>(Ag + kb, RNN_D, sA[nxt]);
    stage_async<256, 2>(Bg + kb, RNN_D, sB[nxt]);
    compute_tile(sA[cur], sB[cur], acc, aRowBase, bRowBase, lane);
    wait_async0();
#else
    v8bf na[2], nb[2];
    stage_gload<256, 2>(Ag + kb, RNN_D, na);
    stage_gload<256, 2>(Bg + kb, RNN_D, nb);
    compute_tile(sA[cur], sB[cur], acc, aRowBase, bRowBase, lane);
    stage_sstore<256, 2>(sA[nxt], na);
    stage_sstore<256, 2>(sB[nxt], nb);
#endif
    __syncthreads();
    cur = nxt;
  }
  compute_tile(sA[cur], sB[cur], acc, aRowBase, bRowBase, lane);

  // C/D layout: VGPR i -> M = i + 8*(lane>=16); N = lane & 15.
  const int rhalf = ((lane >> 4) & 1) * 8;
  const int col   = lane & 15;
#pragma unroll
  for (int mt = 0; mt < 2; ++mt) {
#pragma unroll
    for (int nt = 0; nt < 4; ++nt) {
#pragma unroll
      for (int i = 0; i < 8; ++i) {
        const int grow = mBase + aRowBase + mt * 16 + i + rhalf;
        const int gcol = nBase + bRowBase + nt * 16 + col;
        out[(long long)grow * RNN_H + gcol] = acc[mt][nt][i] + bias[gcol];
      }
    }
  }
}

// ---------------------------------------------------------------------------
// Phase 2 (one launch per timestep t):
//   h_new[b, n] = relu(out[b, t, n] + bias[n] + sum_k hin[b, k] * Wh[n, k])
// Block: 128 threads = 4 waves (2x2), tile 64(M=B) x 128(N), grid = H/128.
// Same double-buffered pipeline; Wh (2 MB bf16) stays L2-resident.
// ---------------------------------------------------------------------------
__global__ __launch_bounds__(128)
void rnn_step_kernel(const unsigned short* __restrict__ hin,   // [B, H] bf16
                     const unsigned short* __restrict__ Whb,   // [H, H] bf16
                     const float* __restrict__ bias,           // [H]
                     float* __restrict__ out,                  // [B, T, H] f32
                     unsigned short* __restrict__ hout,        // [B, H] bf16
                     float* __restrict__ hlast,                // null or [B, H]
                     int t) {
  __shared__ unsigned short sA[2][64 * LDS_PITCH];
  __shared__ unsigned short sB[2][128 * LDS_PITCH];

  const int nBase = blockIdx.x * 128;
  const int wave  = threadIdx.x >> 5;
  const int lane  = threadIdx.x & 31;
  const int waveM = wave >> 1;  // 0..1
  const int waveN = wave & 1;   // 0..1
  const int aRowBase = waveM * 32;
  const int bRowBase = waveN * 64;

  const unsigned short* Ag = hin;
  const unsigned short* Bg = Whb + (long long)nBase * RNN_H;

  v8f acc[2][4] = {};
  constexpr int NK = RNN_H / 32;

#if USE_ASYNC_LDS
  stage_async<128, 2>(Ag, RNN_H, sA[0]);
  stage_async<128, 4>(Bg, RNN_H, sB[0]);
  wait_async0();
#else
  {
    v8bf ra[2], rb[4];
    stage_gload<128, 2>(Ag, RNN_H, ra);
    stage_gload<128, 4>(Bg, RNN_H, rb);
    stage_sstore<128, 2>(sA[0], ra);
    stage_sstore<128, 4>(sB[0], rb);
  }
#endif
  __syncthreads();

  int cur = 0;
  for (int it = 1; it < NK; ++it) {
    const int kb = it * 32;
    const int nxt = cur ^ 1;
#if USE_ASYNC_LDS
    stage_async<128, 2>(Ag + kb, RNN_H, sA[nxt]);
    stage_async<128, 4>(Bg + kb, RNN_H, sB[nxt]);
    compute_tile(sA[cur], sB[cur], acc, aRowBase, bRowBase, lane);
    wait_async0();
#else
    v8bf na[2], nb[4];
    stage_gload<128, 2>(Ag + kb, RNN_H, na);
    stage_gload<128, 4>(Bg + kb, RNN_H, nb);
    compute_tile(sA[cur], sB[cur], acc, aRowBase, bRowBase, lane);
    stage_sstore<128, 2>(sA[nxt], na);
    stage_sstore<128, 4>(sB[nxt], nb);
#endif
    __syncthreads();
    cur = nxt;
  }
  compute_tile(sA[cur], sB[cur], acc, aRowBase, bRowBase, lane);

  const int rhalf = ((lane >> 4) & 1) * 8;
  const int col   = lane & 15;
#pragma unroll
  for (int mt = 0; mt < 2; ++mt) {
#pragma unroll
    for (int nt = 0; nt < 4; ++nt) {
#pragma unroll
      for (int i = 0; i < 8; ++i) {
        const int grow = aRowBase + mt * 16 + i + rhalf;          // b index
        const int gcol = nBase + bRowBase + nt * 16 + col;        // n index
        const long long oidx =
            (long long)grow * (RNN_T * RNN_H) + (long long)t * RNN_H + gcol;
        float v = acc[mt][nt][i] + bias[gcol] + out[oidx];
        v = v > 0.0f ? v : 0.0f;
        out[oidx] = v;
        hout[grow * RNN_H + gcol] = f2bf(v);
        if (hlast) hlast[grow * RNN_H + gcol] = v;
      }
    }
  }
}

// ---------------------------------------------------------------------------
// Host launcher. Workspace layout:
//   Xb  : B*T*D*2 = 64 MiB   bf16 X
//   Web : H*D*2   =  2 MiB   bf16 We
//   Whb : H*H*2   =  2 MiB   bf16 Wh
//   h0,h1 : B*H*2 each       ping-pong bf16 hidden state
// ---------------------------------------------------------------------------
extern "C" void kernel_launch(void* const* d_in, const int* in_sizes, int n_in,
                              void* d_out, int out_size, void* d_ws, size_t ws_size,
                              hipStream_t stream) {
  const float* X    = (const float*)d_in[0];
  const float* We_w = (const float*)d_in[1];
  const float* We_b = (const float*)d_in[2];
  const float* Wh_w = (const float*)d_in[3];
  const float* Wh_b = (const float*)d_in[4];

  float* out   = (float*)d_out;                                  // [B, T, H]
  float* hlast = out + (long long)RNN_B * RNN_T * RNN_H;         // [B, H]

  char* ws = (char*)d_ws;
  unsigned short* Xb  = (unsigned short*)ws;
  unsigned short* Web = Xb  + (size_t)RNN_B * RNN_T * RNN_D;
  unsigned short* Whb = Web + (size_t)RNN_H * RNN_D;
  unsigned short* h0  = Whb + (size_t)RNN_H * RNN_H;
  unsigned short* h1  = h0  + (size_t)RNN_B * RNN_H;

  // 1) f32 -> bf16 conversions + zero h0
  convert_kernel<<<2048, 256, 0, stream>>>(X, We_w, Wh_w, Xb, Web, Whb, h0);

  // 2) Input projection GEMM: xe (+We_b) written f32 into d_out in place
  gemm_xe_kernel<<<dim3((RNN_B * RNN_T) / 128, RNN_H / 128), 256, 0, stream>>>(
      Xb, Web, We_b, out);

  // 3) Sequential recurrence: one launch per timestep, ping-pong h buffers
  for (int t = 0; t < RNN_T; ++t) {
    const unsigned short* hin = (t & 1) ? h1 : h0;
    unsigned short* houtp     = (t & 1) ? h0 : h1;
    rnn_step_kernel<<<RNN_H / 128, 128, 0, stream>>>(
        hin, Whb, Wh_b, out, houtp,
        (t == RNN_T - 1) ? hlast : nullptr, t);
  }
}
